// MPOLinear2_6528350290554
// MI455X (gfx1250) — matrix operand order repaired and checked by reading, not compiled, not run on hardware
//
#include <hip/hip_runtime.h>
#include <hip/hip_bf16.h>

// MPO/TT linear for MI455X (gfx1250, wave32, WMMA bf16 16x16x32).
//
// Fused-core formulation:
//   M1[(w,x),(o,s,p)] = sum_r  first[w,r,o] * mid0[x,r,s,p]   (64 x 2048)
//   M2[(y,s,z),(q,v)] = sum_u  mid1[y,s,u,q] * last[z,u,v]    (2048 x 64)
// Per batch b:
//   GEMM1: t2[(y,z),(o,s,p)] = A1 * M1,  A1[m,k] = x[b, k*64+m]   (64x64x2048)
//   GEMM2: out[(o,p),(q,v)]  = A2 * M2,  A2[(o,p),(y,s,z)] = t2    (64x2048x64)
// t2 lives in LDS (64x2048 bf16 = 256 KB); M1/M2 pre-swizzled into WMMA
// B-fragment order in d_ws by a setup kernel.

typedef __attribute__((ext_vector_type(16))) __bf16 v16bf;
typedef __attribute__((ext_vector_type(8)))  __bf16 v8bf;
typedef __attribute__((ext_vector_type(2)))  __bf16 v2bf;
typedef __attribute__((ext_vector_type(8)))  float  v8f;

union FragU { v16bf v; uint4 q[2]; };

// Native casts -> backend emits v_cvt_pk_bf16_f32 (RNE), not bit-twiddling.
static __device__ __forceinline__ unsigned pack2_bf16(float a, float b) {
    union { v2bf v; unsigned u; } t;
    t.v[0] = (__bf16)a;
    t.v[1] = (__bf16)b;
    return t.u;
}
static __device__ __forceinline__ uint4 pack8_bf16(v8f c) {
    union { v8bf v; uint4 q; } t;
    #pragma unroll
    for (int i = 0; i < 8; ++i) t.v[i] = (__bf16)c[i];
    return t.q;
}

// ---------------------------------------------------------------------------
// Setup: build fused weights directly in per-lane WMMA B-fragment layout.
// B 32x16 bf16 fragment (wave32): lane<16 -> N=lane, K=0..7,16..23 (VGPR j<4: K=2j,2j+1; j>=4: K=2j+8,2j+9)
//                                 lane>=16 -> N=lane-16, K=8..15,24..31.
// M1 frags: [jn(128)][kk(2)][lane(32)][8 dwords]   (dword index == tid bits)
// M2 frags: [ks(64)][nt(4)][lane(32)][8 dwords]
// ---------------------------------------------------------------------------
extern "C" __global__ void tt_setup(const float* __restrict__ fc,
                                    const float* __restrict__ mc0,
                                    const float* __restrict__ mc1,
                                    const float* __restrict__ lc,
                                    unsigned* __restrict__ m1f,
                                    unsigned* __restrict__ m2f) {
    int tid   = blockIdx.x * blockDim.x + threadIdx.x;   // 0..131071
    int d     = tid & 65535;
    int lj    = d & 7;            // dword within lane (K pair)
    int lane  = (d >> 3) & 31;
    int f     = d >> 8;           // fragment id
    int nl    = lane & 15;
    int khalf = lane >> 4;
    if ((tid >> 16) == 0) {
        // M1: K=(w,x) [64], N=(o,s,p) [2048], n = o*256 + s*8 + p
        int kk = f & 1, jn = f >> 1;
        int n = jn * 16 + nl;
        int o = n >> 8, s = (n >> 3) & 31, p = n & 7;
        float v[2];
        #pragma unroll
        for (int h = 0; h < 2; ++h) {
            int k = kk * 32 + (lj & 4) * 4 + khalf * 8 + 2 * (lj & 3) + h;
            int w = k >> 3, xx = k & 7;
            float acc = 0.f;
            for (int r = 0; r < 32; ++r)
                acc += fc[w * 256 + r * 8 + o] * mc0[xx * 8192 + r * 256 + s * 8 + p];
            v[h] = acc;
        }
        m1f[d] = pack2_bf16(v[0], v[1]);
    } else {
        // M2: K=(y,s,z) [2048] k2 = y*256 + s*8 + z, N=(q,v) [64] n2 = q*8 + v
        int nt = f & 3, ks = f >> 2;
        int n2 = nt * 16 + nl;
        int q = n2 >> 3, vv = n2 & 7;
        float v[2];
        #pragma unroll
        for (int h = 0; h < 2; ++h) {
            int k2 = ks * 32 + (lj & 4) * 4 + khalf * 8 + 2 * (lj & 3) + h;
            int y = k2 >> 8, s = (k2 >> 3) & 31, z = k2 & 7;
            float acc = 0.f;
            for (int u = 0; u < 32; ++u)
                acc += mc1[y * 8192 + s * 256 + u * 8 + q] * lc[z * 256 + u * 8 + vv];
            v[h] = acc;
        }
        m2f[d] = pack2_bf16(v[0], v[1]);
    }
}

// ---------------------------------------------------------------------------
// Main kernel: one batch element per block; 256 threads = 8 waves.
// LDS: [0,8K)  xbT  = A1 as bf16 [m=64][k=64]
//      [8K, 8K+256K) t2 = bf16 [m2=64][k2=2048]
// ---------------------------------------------------------------------------
extern "C" __global__ __launch_bounds__(256)
void tt_main(const float* __restrict__ x,
             const unsigned* __restrict__ m1f,
             const unsigned* __restrict__ m2f,
             const float* __restrict__ bias,
             float* __restrict__ out) {
    extern __shared__ char smem[];
    unsigned short* xbT16 = (unsigned short*)smem;
    const int T2OFF = 8192;

    const int b     = blockIdx.x;
    const int tid   = threadIdx.x;
    const int lane  = tid & 31;
    const int wave  = tid >> 5;
    const int nl    = lane & 15;
    const int khalf = lane >> 4;

    // ---- Phase 0: stage x(b) transposed -> bf16 A1[m][k] ----
    const float* xb = x + (size_t)b * 4096;
    #pragma unroll
    for (int j = 0; j < 16; j += 2) {
        int i0 = tid + j * 256;           // global idx: k = i>>6 (w,x), m = i&63 (y,z)
        int i1 = i0 + 256;
        unsigned pk0 = pack2_bf16(xb[i0], xb[i1]);
        xbT16[(i0 & 63) * 64 + (i0 >> 6)] = (unsigned short)(pk0 & 0xFFFFu);
        xbT16[(i1 & 63) * 64 + (i1 >> 6)] = (unsigned short)(pk0 >> 16);
    }
    __syncthreads();

    // ---- Phase A: t2 = A1(64x64) * M1(64x2048), wave owns N-slice o==wave ----
    const uint4* m1q = (const uint4*)m1f;
    for (int mt = 0; mt < 4; ++mt) {
        // A fragments for kk=0,1 (ISA 16-bit A layout: two 8-K runs per lane)
        FragU a0, a1;
        {
            size_t rowb = (size_t)((mt * 16 + nl) * 64) * 2;   // bytes
            a0.q[0] = *(const uint4*)(smem + rowb +       khalf * 16);
            a0.q[1] = *(const uint4*)(smem + rowb + 32  + khalf * 16);
            a1.q[0] = *(const uint4*)(smem + rowb + 64  + khalf * 16);
            a1.q[1] = *(const uint4*)(smem + rowb + 96  + khalf * 16);
        }
        const int y = mt * 2 + khalf;
        for (int j = 0; j < 16; ++j) {
            int jn = wave * 16 + j;
            const uint4* bp0 = m1q + (size_t)((jn * 2 + 0) * 32 + lane) * 2;
            const uint4* bp1 = m1q + (size_t)((jn * 2 + 1) * 32 + lane) * 2;
            FragU b0, b1;
            b0.q[0] = bp0[0]; b0.q[1] = bp0[1];
            b1.q[0] = bp1[0]; b1.q[1] = bp1[1];
            v8f c = {};
            c = __builtin_amdgcn_wmma_f32_16x16x32_bf16(false, a0.v, false, b0.v,
                                                        (short)0, c, false, false);
            c = __builtin_amdgcn_wmma_f32_16x16x32_bf16(false, a1.v, false, b1.v,
                                                        (short)0, c, false, false);
            // epilogue: lane holds z=0..7 at fixed (y,o,s,p) -> contiguous 8 bf16
            int n  = jn * 16 + nl;                 // n = o*256 + s*8 + p
            int o  = n >> 8, s = (n >> 3) & 31, p = n & 7;
            int m2 = o * 8 + p;
            unsigned eo = (unsigned)m2 * 2048u + (unsigned)y * 256u + (unsigned)s * 8u;
            *(uint4*)(smem + T2OFF + (size_t)eo * 2) = pack8_bf16(c);  // ds_store_b128
        }
    }
    __syncthreads();

    // ---- Phase B: out_b = A2(64x2048) * M2(2048x64); wave -> (mt, 2 n-tiles) ----
    const int mt  = wave >> 1;
    const int nt0 = (wave & 1) * 2;
    const uint4* m2q = (const uint4*)m2f;
    v8f c0 = {}, c1 = {};
    size_t arowb = (size_t)T2OFF + (size_t)((mt * 16 + nl) * 2048) * 2;
    for (int ks = 0; ks < 64; ++ks) {
        FragU a;
        a.q[0] = *(const uint4*)(smem + arowb + ks * 64 +      khalf * 16);
        a.q[1] = *(const uint4*)(smem + arowb + ks * 64 + 32 + khalf * 16);
        const uint4* bp0 = m2q + (size_t)((ks * 4 + nt0    ) * 32 + lane) * 2;
        const uint4* bp1 = m2q + (size_t)((ks * 4 + nt0 + 1) * 32 + lane) * 2;
        FragU b0, b1;
        b0.q[0] = bp0[0]; b0.q[1] = bp0[1];
        b1.q[0] = bp1[0]; b1.q[1] = bp1[1];
        c0 = __builtin_amdgcn_wmma_f32_16x16x32_bf16(false, a.v, false, b0.v,
                                                     (short)0, c0, false, false);
        c1 = __builtin_amdgcn_wmma_f32_16x16x32_bf16(false, a.v, false, b1.v,
                                                     (short)0, c1, false, false);
    }
    // ---- epilogue: bias add + store (out layout (o,p,q,v) = m2*64 + n2) ----
    float* ob = out + (size_t)b * 4096;
    const int n2a = nt0 * 16 + nl;
    const int n2b = (nt0 + 1) * 16 + nl;
    #pragma unroll
    for (int i = 0; i < 8; ++i) {
        int m2 = mt * 16 + khalf * 8 + i;
        ob[m2 * 64 + n2a] = c0[i] + bias[m2 * 64 + n2a];
        ob[m2 * 64 + n2b] = c1[i] + bias[m2 * 64 + n2b];
    }
}

extern "C" void kernel_launch(void* const* d_in, const int* in_sizes, int n_in,
                              void* d_out, int out_size, void* d_ws, size_t ws_size,
                              hipStream_t stream) {
    (void)in_sizes; (void)n_in; (void)out_size; (void)ws_size;
    const float* x    = (const float*)d_in[0];
    const float* fc   = (const float*)d_in[1];
    const float* mc0  = (const float*)d_in[2];
    const float* mc1  = (const float*)d_in[3];
    const float* lc   = (const float*)d_in[4];
    const float* bias = (const float*)d_in[5];

    unsigned* m1f = (unsigned*)d_ws;          // 256 KB
    unsigned* m2f = m1f + 65536;              // 256 KB

    tt_setup<<<512, 256, 0, stream>>>(fc, mc0, mc1, lc, m1f, m2f);

    size_t lds_bytes = 8192 + 64 * 2048 * 2;  // xbT + t2 = 264 KB (<= 320 KB/WGP)
    tt_main<<<2048, 256, lds_bytes, stream>>>(x, m1f, m2f, bias, (float*)d_out);
}
